// CycleVAR_21182778704231
// MI455X (gfx1250) — matrix operand order repaired and checked by reading, not compiled, not run on hardware
//
#include <hip/hip_runtime.h>

// ---------------------------------------------------------------------------
// CycleVAR forward for MI455X (gfx1250, wave32, WMMA).
// Core compute: nearest-codebook search via v_wmma_f32_16x16x32_bf16
// (K=32 == channel dim -> exactly one WMMA per 16x16 token*code tile).
// Quantizer loop is 3-stage software-pipelined:
//   stage 0: B tiles prefetched one iteration ahead (global loads)
//   stage 1: WMMA pair issued on current tiles (distinct live dest regs)
//   stage 2: argmax-consume of the PREVIOUS iteration's accumulators
// so the BF16 WMMA->VALU 4-coexec hazard and L2 latency are both hidden.
// ---------------------------------------------------------------------------

#define B_   128
#define C_   32
#define H_   16
#define V_   4096
#define NSC  10
#define TOT_TOK 680

__constant__ int d_PN[NSC] = {1, 2, 3, 4, 5, 6, 8, 10, 13, 16};

typedef __attribute__((ext_vector_type(16))) __bf16 v16bf;
typedef __attribute__((ext_vector_type(8)))  float  v8f;

// ---------------- helpers ----------------

__device__ __forceinline__ unsigned short f2bf_bits(float f) {
    unsigned int u = __float_as_uint(f);
    unsigned int r = (u + 0x7FFFu + ((u >> 16) & 1u)) >> 16;   // RNE
    return (unsigned short)r;
}

__device__ __forceinline__ float cubicw(float t) {
    t = fabsf(t);
    const float a = -0.75f;
    if (t <= 1.f) return ((a + 2.f) * t - (a + 3.f)) * t * t + 1.f;
    if (t <  2.f) return (((t - 5.f) * t + 8.f) * t - 4.f) * a;
    return 0.f;
}

// ---------------- setup: bf16 codebook, 0.5*|c|^2, bicubic matrices --------

__global__ void setup_kernel(const float* __restrict__ codebook,
                             unsigned short* __restrict__ cb_bf,
                             float* __restrict__ c2h,
                             float* __restrict__ U) {
    int tid = blockIdx.x * blockDim.x + threadIdx.x;
    int nth = gridDim.x * blockDim.x;
    // codebook fp32 -> bf16 bits
    for (int i = tid; i < V_ * C_; i += nth)
        cb_bf[i] = f2bf_bits(codebook[i]);
    // 0.5 * |c|^2 per code  (argmax(2S - |c|^2) == argmax(S - 0.5|c|^2))
    if (tid < V_) {
        const float* p = codebook + tid * C_;
        float s = 0.f;
        #pragma unroll
        for (int k = 0; k < C_; k++) s += p[k] * p[k];
        c2h[tid] = 0.5f * s;
    }
    // bicubic upsample matrices U[si][16][16] (pn -> 16); identity for pn==16
    if (blockIdx.x == 0 && threadIdx.x < NSC * 16) {
        int si = threadIdx.x >> 4;
        int j  = threadIdx.x & 15;
        int pn = d_PN[si];
        float* row = U + si * 256 + j * 16;
        for (int k = 0; k < 16; k++) row[k] = 0.f;
        float scale = (float)pn / 16.f;
        float src = ((float)j + 0.5f) * scale - 0.5f;
        float fi  = floorf(src);
        float t   = src - fi;
        int   fii = (int)fi;
        #pragma unroll
        for (int k = -1; k < 3; k++) {
            int idx = fii + k;
            idx = idx < 0 ? 0 : (idx > pn - 1 ? pn - 1 : idx);
            row[idx] += cubicw((float)k - t);
        }
    }
}

__global__ void init_kernel(const float* __restrict__ feat,
                            float* __restrict__ f_rest,
                            float* __restrict__ f_hat, int n) {
    for (int i = blockIdx.x * blockDim.x + threadIdx.x; i < n;
         i += gridDim.x * blockDim.x) {
        f_rest[i] = feat[i];
        f_hat[i]  = 0.f;
    }
}

// ---------------- area downsample: [B,C,16,16] -> token-major [*,C] --------

__global__ void area_down(const float* __restrict__ src,
                          float* __restrict__ dst,
                          int pn, int bstride, int tokoff) {
    int i = blockIdx.x * blockDim.x + threadIdx.x;
    int total = B_ * C_ * pn * pn;
    if (i >= total) return;
    int j  = i % pn;
    int ii = (i / pn) % pn;
    int c  = (i / (pn * pn)) % C_;
    int b  = i / (pn * pn * C_);
    int ys = (ii * H_) / pn, ye = ((ii + 1) * H_ + pn - 1) / pn;
    int xs = (j  * H_) / pn, xe = ((j  + 1) * H_ + pn - 1) / pn;
    const float* p = src + ((b * C_ + c) * H_) * H_;
    float acc = 0.f;
    for (int y = ys; y < ye; y++)
        for (int x = xs; x < xe; x++)
            acc += p[y * H_ + x];
    acc *= 1.f / (float)((ye - ys) * (xe - xs));
    dst[b * bstride + (tokoff + ii * pn + j) * C_ + c] = acc;
}

// ---------------- WMMA nearest-codebook quantizer --------------------------

__global__ void __launch_bounds__(256)
quantize_wmma(const float* __restrict__ z_tok,
              const unsigned short* __restrict__ cb_bits,
              const float* __restrict__ c2h,
              const float* __restrict__ codebook,
              float* __restrict__ h_tok, int n_tok) {
    const int lane = threadIdx.x & 31;
    const int wave = threadIdx.x >> 5;
    const int m0   = (blockIdx.x * (blockDim.x >> 5) + wave) * 16;
    if (m0 >= n_tok) return;   // uniform per wave

    const __bf16* cb = reinterpret_cast<const __bf16*>(cb_bits);

    // A tile (16 tokens x K=32 bf16), contiguous K order matching B:
    // lanes 0-15: row=lane, K {0..15}; lanes 16-31: row=lane-16, K {16..31}
    int row   = m0 + (lane & 15);
    int kbase = (lane < 16) ? 0 : 16;
    v16bf a;
    if (row < n_tok) {
        const float* zp = z_tok + row * C_;
        #pragma unroll
        for (int e = 0; e < 8; e++) {
            a[e]     = (__bf16)zp[kbase + e];
            a[8 + e] = (__bf16)zp[kbase + 8 + e];
        }
    } else {
        #pragma unroll
        for (int e = 0; e < 16; e++) a[e] = (__bf16)0.f;
    }

    const int col  = lane & 15;            // C/D: N = lane % 16
    const int koff = (lane < 16) ? 0 : 16; // B: lane n -> K 0-15, lane n+16 -> K 16-31

    float bestv[8];
    int   besti[8];
    #pragma unroll
    for (int r = 0; r < 8; r++) { bestv[r] = -3.4e38f; besti[r] = 0; }

    auto loadB = [&](int v0) -> v16bf {
        return *reinterpret_cast<const v16bf*>(cb + (v0 + col) * C_ + koff);
    };
    auto domma = [&](const v16bf& b) -> v8f {
        v8f z = {0.f, 0.f, 0.f, 0.f, 0.f, 0.f, 0.f, 0.f};
        return __builtin_amdgcn_wmma_f32_16x16x32_bf16(
            false, a, false, b, (short)0, z, false, false);
    };
    auto consume = [&](const v8f& acc, int v0, float c2v) {
        #pragma unroll
        for (int r = 0; r < 8; r++) {
            float val = acc[r] - c2v;
            if (val > bestv[r]) { bestv[r] = val; besti[r] = v0 + col; }
        }
    };

    // ---- 3-stage pipeline ----
    // current tiles {0,16}; next tiles {32,48}
    v16bf cur0 = loadB(0),  cur1 = loadB(16);
    v16bf nxt0 = loadB(32), nxt1 = loadB(48);
    float pc0 = c2h[col],       pc1 = c2h[16 + col];
    float nc0 = c2h[32 + col],  nc1 = c2h[48 + col];

    v8f accA = domma(cur0);
    v8f accB = domma(cur1);
    int pend = 0;

    for (int v0 = 32; v0 < V_; v0 += 32) {
        // stage 0: prefetch tiles {v0+32, v0+48} (masked wrap on last iter)
        int pf = (v0 + 32) & (V_ - 1);
        v16bf pf0 = loadB(pf), pf1 = loadB(pf + 16);
        float pfc0 = c2h[pf + col], pfc1 = c2h[pf + 16 + col];

        // stage 1: WMMA pair on current tiles (distinct dest regs, in flight)
        v8f nA = domma(nxt0);
        v8f nB = domma(nxt1);

        // stage 2: consume previous iteration's accumulators
        consume(accA, pend, pc0);
        consume(accB, pend + 16, pc1);

        accA = nA;  accB = nB;
        pend = v0;  pc0 = nc0;  pc1 = nc1;
        nxt0 = pf0; nxt1 = pf1; nc0 = pfc0; nc1 = pfc1;
    }
    consume(accA, pend, pc0);
    consume(accB, pend + 16, pc1);

    // argmax across the 16 lanes sharing each row (half-wave groups)
    #pragma unroll
    for (int r = 0; r < 8; r++) {
        float v = bestv[r];
        int   ix = besti[r];
        #pragma unroll
        for (int off = 1; off < 16; off <<= 1) {
            float ov = __shfl_xor(v, off, 16);
            int   oi = __shfl_xor(ix, off, 16);
            if (ov > v || (ov == v && oi < ix)) { v = ov; ix = oi; }
        }
        besti[r] = ix;
    }

    // gather winning fp32 codebook rows: each 16-lane group writes its 8 rows,
    // each lane writes 2 channels.
    int rbase = (lane < 16) ? 0 : 8;
    #pragma unroll
    for (int r = 0; r < 8; r++) {
        int rr = m0 + rbase + r;
        if (rr < n_tok) {
            const float* cw = codebook + besti[r] * C_;
            h_tok[rr * C_ + 2 * col]     = cw[2 * col];
            h_tok[rr * C_ + 2 * col + 1] = cw[2 * col + 1];
        }
    }
}

// ---------------- separable bicubic upsample pn -> 16 ----------------------

__global__ void up_v(const float* __restrict__ h_tok,
                     const float* __restrict__ U,
                     float* __restrict__ tmp, int pn) {
    int i = blockIdx.x * blockDim.x + threadIdx.x;
    int total = B_ * C_ * H_ * pn;
    if (i >= total) return;
    int px = i % pn;
    int y  = (i / pn) % H_;
    int c  = (i / (pn * H_)) % C_;
    int b  = i / (pn * H_ * C_);
    float acc = 0.f;
    for (int py = 0; py < pn; py++)
        acc += U[y * 16 + py] * h_tok[(b * pn * pn + py * pn + px) * C_ + c];
    tmp[((b * C_ + c) * H_ + y) * pn + px] = acc;
}

__global__ void up_h(const float* __restrict__ tmp,
                     const float* __restrict__ U,
                     float* __restrict__ h_up, int pn) {
    int i = blockIdx.x * blockDim.x + threadIdx.x;
    if (i >= B_ * C_ * H_ * H_) return;
    int x = i % H_;
    int y = (i / H_) % H_;
    int c = (i / (H_ * H_)) % C_;
    int b = i / (H_ * H_ * C_);
    float acc = 0.f;
    for (int px = 0; px < pn; px++)
        acc += U[x * 16 + px] * tmp[((b * C_ + c) * H_ + y) * pn + px];
    h_up[i] = acc;   // layout [b][c][y][x]
}

// ---------------- 3x3 conv + blend + residual update -----------------------

__global__ void conv_blend(const float* __restrict__ h_up,
                           const float* __restrict__ w,    // [32][32][3][3]
                           const float* __restrict__ bias, // [32]
                           float* __restrict__ f_hat,
                           float* __restrict__ f_rest) {
    __shared__ float s_in[C_][18][18];   // zero-padded halo
    const int b = blockIdx.x;
    const int tid = threadIdx.x;         // 256 threads
    float* s_flat = &s_in[0][0][0];
    for (int i = tid; i < C_ * 18 * 18; i += 256) s_flat[i] = 0.f;
    __syncthreads();
    const float* in = h_up + b * C_ * H_ * H_;
    for (int i = tid; i < C_ * H_ * H_; i += 256) {
        int c = i >> 8, y = (i >> 4) & 15, x = i & 15;
        s_in[c][y + 1][x + 1] = in[i];
    }
    __syncthreads();
    const int x = tid & 15, y = (tid >> 4) & 15;
    for (int co = 0; co < C_; co++) {
        float acc = bias[co];
        const float* wco = w + co * C_ * 9;
        #pragma unroll 4
        for (int ci = 0; ci < C_; ci++) {
            const float* w9 = wco + ci * 9;
            const float* sp = &s_in[ci][y][x];
            acc += w9[0]*sp[0]  + w9[1]*sp[1]  + w9[2]*sp[2]
                 + w9[3]*sp[18] + w9[4]*sp[19] + w9[5]*sp[20]
                 + w9[6]*sp[36] + w9[7]*sp[37] + w9[8]*sp[38];
        }
        float hv   = s_in[co][y + 1][x + 1];
        float outv = 0.5f * hv + 0.5f * acc;     // RESI_RATIO = 0.5
        int idx = ((b * C_ + co) * H_ + y) * H_ + x;
        f_hat[idx]  += outv;
        f_rest[idx] -= outv;
    }
}

// ---------------- launcher -------------------------------------------------

extern "C" void kernel_launch(void* const* d_in, const int* in_sizes, int n_in,
                              void* d_out, int out_size, void* d_ws, size_t ws_size,
                              hipStream_t stream) {
    (void)in_sizes; (void)n_in; (void)out_size; (void)ws_size;
    const float* feat     = (const float*)d_in[0];  // [128,32,16,16]
    const float* codebook = (const float*)d_in[1];  // [4096,32]
    const float* phi_w    = (const float*)d_in[2];  // [4,32,32,3,3]
    const float* phi_b    = (const float*)d_in[3];  // [4,32]
    float* out = (float*)d_out;                     // [128,680,32]

    static const int PN[NSC]      = {1, 2, 3, 4, 5, 6, 8, 10, 13, 16};
    static const int TOK_OFF[NSC] = {0, 1, 5, 14, 30, 55, 91, 155, 255, 424};
    static const int PHI_IDX[NSC] = {0, 0, 1, 1, 1, 2, 2, 2, 3, 3};

    const size_t NFEAT = (size_t)B_ * C_ * H_ * H_;   // 1,048,576 floats
    char* ws = (char*)d_ws;
    size_t o = 0;
    auto alloc = [&](size_t bytes) {
        void* p = ws + o;
        o += (bytes + 255) & ~(size_t)255;
        return p;
    };
    float* f_rest       = (float*)alloc(NFEAT * 4);
    float* f_hat        = (float*)alloc(NFEAT * 4);
    float* z_tok        = (float*)alloc(NFEAT * 4);
    float* h_tok        = (float*)alloc(NFEAT * 4);
    float* h_up         = (float*)alloc(NFEAT * 4);
    float* tmp          = (float*)alloc(NFEAT * 4);
    unsigned short* cbb = (unsigned short*)alloc((size_t)V_ * C_ * 2);
    float* c2h          = (float*)alloc(V_ * 4);
    float* U            = (float*)alloc(NSC * 256 * 4);

    setup_kernel<<<64, 256, 0, stream>>>(codebook, cbb, c2h, U);
    init_kernel<<<1024, 256, 0, stream>>>(feat, f_rest, f_hat, (int)NFEAT);

    for (int si = 0; si < NSC; si++) {
        const int pn = PN[si];
        const int n_tok = B_ * pn * pn;

        // z = area-downsample(f_rest) -> token-major [n_tok, 32]
        {
            int total = B_ * C_ * pn * pn;
            area_down<<<(total + 255) / 256, 256, 0, stream>>>(
                f_rest, z_tok, pn, pn * pn * C_, 0);
        }
        // nearest-codebook gather (forward value of the STE)
        {
            int tiles = (n_tok + 15) / 16;
            quantize_wmma<<<(tiles + 7) / 8, 256, 0, stream>>>(
                z_tok, cbb, c2h, codebook, h_tok, n_tok);
        }
        // bicubic upsample to 16x16 (identity matrix for pn==16)
        {
            int t1 = B_ * C_ * H_ * pn;
            up_v<<<(t1 + 255) / 256, 256, 0, stream>>>(h_tok, U + si * 256, tmp, pn);
            int t2 = B_ * C_ * H_ * H_;
            up_h<<<(t2 + 255) / 256, 256, 0, stream>>>(tmp, U + si * 256, h_up, pn);
        }
        // phi conv + blend, update f_hat / f_rest
        conv_blend<<<B_, 256, 0, stream>>>(
            h_up, phi_w + (size_t)PHI_IDX[si] * C_ * C_ * 9,
            phi_b + PHI_IDX[si] * C_, f_hat, f_rest);
        // context tokens: area-downsample(f_hat) into output slice
        {
            int total = B_ * C_ * pn * pn;
            area_down<<<(total + 255) / 256, 256, 0, stream>>>(
                f_hat, out, pn, TOT_TOK * C_, TOK_OFF[si]);
        }
    }
}
